// CausalSelfAttention_73744588473038
// MI455X (gfx1250) — matrix operand order repaired and checked by reading, not compiled, not run on hardware
//
#include <hip/hip_runtime.h>
#include <hip/hip_bf16.h>
#include <math.h>

// ---------------------------------------------------------------------------
// CausalSelfAttention (GQA + RoPE + fake-quant + FIRE bias) for MI455X gfx1250
// All matmuls via v_wmma_f32_16x16x32_bf16 (wave32).
// Attention: 4-wave workgroups sharing one 16-row q-tile score buffer in LDS.
// ---------------------------------------------------------------------------

#define TSEQ  2048
#define CDIM  1024
#define NHEAD 16
#define NKV   4
#define HS    64
#define KVDIM 256

typedef __attribute__((ext_vector_type(16))) __bf16 v16bf;
typedef __attribute__((ext_vector_type(8)))  float  v8f;

union FragU { v16bf v; uint4 q[2]; };

__device__ inline unsigned short f2bf(float f) {
  unsigned int u = __float_as_uint(f);
  unsigned int r = u + 0x7FFFu + ((u >> 16) & 1u);   // round-to-nearest-even
  return (unsigned short)(r >> 16);
}

// A-matrix 16x32 bf16 fragment (ISA 7.12.2): lanes 0-15 row M, K=0..7 & 16..23;
// lanes 16-31 same rows, K=8..15 & 24..31.  Source row-major (row, K), ld elems.
__device__ inline v16bf load_a_frag(const unsigned short* base, int ld) {
  int lane = threadIdx.x & 31;
  int row  = lane & 15;
  int koff = (lane >> 4) << 3;                 // 0 or 8
  const unsigned short* p = base + (size_t)row * ld + koff;
  FragU f;
  f.q[0] = *(const uint4*)(p);                 // K = koff .. koff+7
  f.q[1] = *(const uint4*)(p + 16);            // K = koff+16 .. koff+23
  return f.v;
}

// B-matrix 32x16 bf16 fragment: lanes 0-15 hold K=0..7 & 16..23 of column N=lane,
// lanes 16-31 hold K=8..15 & 24..31.  Source is B^T row-major (row = N, col = K).
__device__ inline v16bf load_b_frag(const unsigned short* base, int ld) {
  int lane = threadIdx.x & 31;
  int col  = lane & 15;
  int koff = (lane >> 4) << 3;                 // 0 or 8
  const unsigned short* p = base + (size_t)col * ld + koff;
  FragU f;
  f.q[0] = *(const uint4*)(p);                 // K = koff .. koff+7
  f.q[1] = *(const uint4*)(p + 16);            // K = koff+16 .. koff+23
  return f.v;
}

// C/D 16x16 f32: VGPR r holds (M = r + 8*(lane/16), N = lane%16)
__device__ inline void store_cd(float* base, int ld, v8f c) {
  int lane = threadIdx.x & 31;
  int nn = lane & 15;
  int mb = (lane >> 4) << 3;
#pragma unroll
  for (int r = 0; r < 8; ++r) base[(size_t)(mb + r) * ld + nn] = c[r];
}

__device__ inline v8f wmma_bf16(v16bf a, v16bf b, v8f c) {
  return __builtin_amdgcn_wmma_f32_16x16x32_bf16(false, a, false, b, (short)0, c,
                                                 false, false);
}

// order-preserving float <-> uint for atomic min/max
__device__ inline unsigned int fenc(float f) {
  unsigned int u = __float_as_uint(f);
  return (u & 0x80000000u) ? ~u : (u | 0x80000000u);
}
__device__ inline float fdec(unsigned int e) {
  unsigned int u = (e & 0x80000000u) ? (e & 0x7FFFFFFFu) : ~e;
  return __uint_as_float(u);
}

// per-tensor affine quant-dequant (BITS=8)
__device__ inline float qdq(float x, float mn, float mx) {
  float scale = (mx - mn) * (1.0f / 255.0f) + 1e-12f;
  float zp = rintf(-mn / scale);
  float q = fminf(fmaxf(rintf(x / scale) + zp, 0.0f), 255.0f);
  return (q - zp) * scale;
}

// ---------------------------------------------------------------------------
// Elementwise / conversion kernels
// ---------------------------------------------------------------------------
__global__ void f32_to_bf16(const float* __restrict__ src,
                            unsigned short* __restrict__ dst, int n) {
  int i = blockIdx.x * blockDim.x + threadIdx.x;
  if (i < n) dst[i] = f2bf(src[i]);
}

// W (K,N) row-major f32 -> Wt (N,K) row-major bf16
__global__ void transpose_f32_to_bf16(const float* __restrict__ W,
                                      unsigned short* __restrict__ Wt, int K, int N) {
  int i = blockIdx.x * blockDim.x + threadIdx.x;
  if (i < K * N) {
    int n = i / K, k = i - n * K;
    Wt[i] = f2bf(W[(size_t)k * N + n]);
  }
}

// in-place rotate-half RoPE on (T, nh, 64) f32
__global__ void rope_kernel(float* __restrict__ p, int nh) {
  int i = blockIdx.x * blockDim.x + threadIdx.x;
  int half = i & 31;
  int head = (i >> 5) % nh;
  int t    = i / (nh * 32);
  if (t >= TSEQ) return;
  float inv = powf(10000.0f, -(float)half * (1.0f / 32.0f));
  float ang = (float)t * inv;
  float cs = cosf(ang), sn = sinf(ang);
  float* row = p + ((size_t)t * nh + head) * HS;
  float x1 = row[half], x2 = row[half + 32];
  row[half]      = x1 * cs - x2 * sn;
  row[half + 32] = x2 * cs + x1 * sn;
}

__global__ void init_mm(unsigned int* mm) {
  int i = threadIdx.x;
  if (i < 8) mm[i] = (i & 1) ? 0u : 0xFFFFFFFFu;
}

__global__ void __launch_bounds__(256)
minmax_kernel(const float* __restrict__ p, int n, unsigned int* __restrict__ mm) {
  __shared__ unsigned int smn[256], smx[256];
  unsigned int mn = 0xFFFFFFFFu, mx = 0u;
  for (int i = blockIdx.x * blockDim.x + threadIdx.x; i < n;
       i += gridDim.x * blockDim.x) {
    unsigned int e = fenc(p[i]);
    mn = min(mn, e); mx = max(mx, e);
  }
  smn[threadIdx.x] = mn; smx[threadIdx.x] = mx;
  __syncthreads();
  for (int s = 128; s > 0; s >>= 1) {
    if ((int)threadIdx.x < s) {
      smn[threadIdx.x] = min(smn[threadIdx.x], smn[threadIdx.x + s]);
      smx[threadIdx.x] = max(smx[threadIdx.x], smx[threadIdx.x + s]);
    }
    __syncthreads();
  }
  if (threadIdx.x == 0) { atomicMin(mm, smn[0]); atomicMax(mm + 1, smx[0]); }
}

// src (T, nh*64) -> dst [head][T][64], quant-dequant -> bf16
__global__ void quant_qk(const float* __restrict__ src,
                         unsigned short* __restrict__ dst,
                         const unsigned int* __restrict__ mm, int mmoff, int nh) {
  int i = blockIdx.x * blockDim.x + threadIdx.x;
  if (i >= TSEQ * nh * HS) return;
  float mn = fdec(mm[mmoff]), mx = fdec(mm[mmoff + 1]);
  int d = i & 63;
  int head = (i >> 6) % nh;
  int t = i / (nh * HS);
  dst[((size_t)head * TSEQ + t) * HS + d] = f2bf(qdq(src[i], mn, mx));
}

// src (T, 256) -> dst [g][d][T]  (V transposed for PV B-fragments)
__global__ void quant_v(const float* __restrict__ src,
                        unsigned short* __restrict__ dst,
                        const unsigned int* __restrict__ mm) {
  int i = blockIdx.x * blockDim.x + threadIdx.x;
  if (i >= TSEQ * KVDIM) return;
  float mn = fdec(mm[4]), mx = fdec(mm[5]);
  int d = i & 63;
  int g = (i >> 6) & 3;
  int t = i >> 8;
  dst[((size_t)g * HS + d) * TSEQ + t] = f2bf(qdq(src[i], mn, mx));
}

// src [h][T][64] -> dst (T, 1024) bf16
__global__ void quant_y(const float* __restrict__ src,
                        unsigned short* __restrict__ dst,
                        const unsigned int* __restrict__ mm) {
  int i = blockIdx.x * blockDim.x + threadIdx.x;
  if (i >= NHEAD * TSEQ * HS) return;
  float mn = fdec(mm[6]), mx = fdec(mm[7]);
  int d = i & 63;
  int t = (i >> 6) & (TSEQ - 1);
  int h = i >> 17;                              // / (TSEQ*HS)
  dst[(size_t)t * CDIM + h * HS + d] = f2bf(qdq(src[i], mn, mx));
}

// ---------------------------------------------------------------------------
// WMMA GEMM: C (M,N) f32 = A (M,K) bf16 @ Bt^T, Bt stored (N,K) row-major bf16.
// One wave computes a 16x64 tile (A-fragment reused across 4 B-fragments).
// ---------------------------------------------------------------------------
__global__ void __launch_bounds__(256)
gemm_bf16_nt(const unsigned short* __restrict__ A,
             const unsigned short* __restrict__ Bt,
             float* __restrict__ C, int M, int N, int K) {
  int wid = (blockIdx.x * blockDim.x + threadIdx.x) >> 5;
  int nt4 = N >> 6;
  int mt  = wid / nt4;
  int n0  = (wid % nt4) << 6;
  if (mt >= (M >> 4)) return;                  // uniform per wave
  const unsigned short* a = A + (size_t)mt * 16 * K;
  v8f acc0 = {}, acc1 = {}, acc2 = {}, acc3 = {};
  for (int k = 0; k < K; k += 32) {
    v16bf af = load_a_frag(a + k, K);
    __builtin_prefetch(a + k + 64, 0, 3);      // global_prefetch_b8 (near cache)
    const unsigned short* b = Bt + (size_t)n0 * K + k;
    acc0 = wmma_bf16(af, load_b_frag(b,                  K), acc0);
    acc1 = wmma_bf16(af, load_b_frag(b + (size_t)16 * K, K), acc1);
    acc2 = wmma_bf16(af, load_b_frag(b + (size_t)32 * K, K), acc2);
    acc3 = wmma_bf16(af, load_b_frag(b + (size_t)48 * K, K), acc3);
  }
  float* cb = C + (size_t)mt * 16 * N + n0;
  store_cd(cb,      N, acc0);
  store_cd(cb + 16, N, acc1);
  store_cd(cb + 32, N, acc2);
  store_cd(cb + 48, N, acc3);
}

// ---------------------------------------------------------------------------
// Fused attention: one 4-wave workgroup = one head x one 16-row q-tile.
// LDS layout (dynamic):
//   S    [16][TSEQ] f32   : masked scores + FIRE bias      (131072 B)
//   obuf [16][64]  f32    : output-tile merge buffer        (  4096 B)
//   Pst  [4][16][32] bf16 : per-wave P staging              (  4096 B)
//   fire [32] float4      : {w1, b1, w2[:,h], 0}            (   512 B)
//   mred [4][16] f32, lred [4][16] f32                      (   512 B)
// Waves split the k-range round-robin; row max / row sum / O merged via LDS.
// ---------------------------------------------------------------------------
#define SMEM_S     0
#define SMEM_OBUF  (16 * TSEQ * 4)
#define SMEM_PST   (SMEM_OBUF + 16 * HS * 4)
#define SMEM_FIRE  (SMEM_PST + 4 * 16 * 32 * 2)
#define SMEM_MRED  (SMEM_FIRE + 32 * 16)
#define SMEM_LRED  (SMEM_MRED + 4 * 16 * 4)
#define SMEM_TOTAL (SMEM_LRED + 4 * 16 * 4)

__global__ void __launch_bounds__(128)
attn_kernel(const unsigned short* __restrict__ qb,
            const unsigned short* __restrict__ kb,
            const unsigned short* __restrict__ vtb,
            float* __restrict__ yf,
            const float* __restrict__ fw1, const float* __restrict__ fb1,
            const float* __restrict__ fw2, const float* __restrict__ fb2,
            const float* __restrict__ fc,  const float* __restrict__ fL) {
  extern __shared__ char smem[];
  float* S             = (float*)(smem + SMEM_S);
  float* obuf          = (float*)(smem + SMEM_OBUF);
  unsigned short* PstW = (unsigned short*)(smem + SMEM_PST);
  float4* fire         = (float4*)(smem + SMEM_FIRE);
  float* mred          = (float*)(smem + SMEM_MRED);
  float* lred          = (float*)(smem + SMEM_LRED);

  const int qt   = blockIdx.x;
  const int h    = blockIdx.y;
  const int g    = h >> 2;                                   // H/G = 4
  const int tid  = threadIdx.x;
  const int w    = tid >> 5;                                 // wave 0..3
  const int lane = tid & 31;
  const int nn   = lane & 15;
  const int mb   = (lane >> 4) << 3;
  unsigned short* Pst = PstW + w * 512;                      // 16x32 bf16

  // init: FIRE weights interleaved as float4, zero output merge buffer
  if (tid < 32)
    fire[tid] = make_float4(fw1[tid], fb1[tid], fw2[tid * NHEAD + h], 0.0f);
  float4 z4 = make_float4(0.0f, 0.0f, 0.0f, 0.0f);
  ((float4*)obuf)[tid]       = z4;
  ((float4*)obuf)[tid + 128] = z4;
  const float b2h  = fb2[h];
  const float cc   = fc[0];
  const float Labs = fabsf(fL[0]);
  __syncthreads();

  float pninv[8];
#pragma unroll
  for (int r = 0; r < 8; ++r) {
    float qp = (float)(qt * 16 + mb + r);
    pninv[r] = 1.0f / (logf(fabsf(cc * fmaxf(qp, Labs)) + 1.0f) + 1e-6f);
  }

  const unsigned short* qbase = qb + ((size_t)h * TSEQ + qt * 16) * HS;
  v16bf a0 = load_a_frag(qbase,      HS);                    // d 0..31
  v16bf a1 = load_a_frag(qbase + 32, HS);                    // d 32..63

  float m[8];
#pragma unroll
  for (int r = 0; r < 8; ++r) m[r] = -3.0e38f;

  // ---- Phase 1: scores + FIRE bias + causal mask (k-tiles kt == w mod 4) ----
  for (int kt = w; kt <= qt; kt += 4) {
    const unsigned short* kbase = kb + ((size_t)g * TSEQ + kt * 16) * HS;
    v8f s = {};
    s = wmma_bf16(a0, load_b_frag(kbase,      HS), s);
    s = wmma_bf16(a1, load_b_frag(kbase + 32, HS), s);
    const int kpos = kt * 16 + nn;
    // FIRE MLP: j-loop hoisted over the 8 rows so each ds_load_b128 broadcast
    // is amortized 8x
    float nd8[8], acc8[8];
#pragma unroll
    for (int r = 0; r < 8; ++r) {
      const int qpos = qt * 16 + mb + r;
      nd8[r] = logf(fabsf(cc * (float)(qpos - kpos)) + 1.0f) * pninv[r];
      acc8[r] = b2h;
    }
#pragma unroll 4
    for (int j = 0; j < 32; ++j) {
      float4 f = fire[j];
#pragma unroll
      for (int r = 0; r < 8; ++r)
        acc8[r] += fmaxf(nd8[r] * f.x + f.y, 0.0f) * f.z;
    }
#pragma unroll
    for (int r = 0; r < 8; ++r) {
      const int qpos = qt * 16 + mb + r;
      float sv = (kpos <= qpos) ? (s[r] * 0.125f + acc8[r]) : -3.0e38f;
      S[(mb + r) * TSEQ + kpos] = sv;
      float v = sv;
#pragma unroll
      for (int off = 1; off < 16; off <<= 1)
        v = fmaxf(v, __shfl_xor(v, off, 32));
      m[r] = fmaxf(m[r], v);
    }
  }
  if (nn == 0) {
#pragma unroll
    for (int r = 0; r < 8; ++r) mred[w * 16 + mb + r] = m[r];
  }
  __syncthreads();
#pragma unroll
  for (int r = 0; r < 8; ++r) {
    float mv = mred[mb + r];
    mv = fmaxf(mv, mred[16 + mb + r]);
    mv = fmaxf(mv, mred[32 + mb + r]);
    mv = fmaxf(mv, mred[48 + mb + r]);
    m[r] = mv;
  }

  // ---- Phase 2: row sums of exp(S - m) over own k-tiles ----
  float l[8];
#pragma unroll
  for (int r = 0; r < 8; ++r) l[r] = 0.0f;
  for (int kt = w; kt <= qt; kt += 4) {
    const int col = kt * 16 + nn;
#pragma unroll
    for (int r = 0; r < 8; ++r)
      l[r] += __expf(S[(mb + r) * TSEQ + col] - m[r]);
  }
#pragma unroll
  for (int r = 0; r < 8; ++r) {
#pragma unroll
    for (int off = 1; off < 16; off <<= 1) l[r] += __shfl_xor(l[r], off, 32);
  }
  if (nn == 0) {
#pragma unroll
    for (int r = 0; r < 8; ++r) lred[w * 16 + mb + r] = l[r];
  }
  __syncthreads();
  float linv[8];
#pragma unroll
  for (int r = 0; r < 8; ++r) {
    float lv = lred[mb + r] + lred[16 + mb + r] + lred[32 + mb + r] +
               lred[48 + mb + r];
    linv[r] = 1.0f / lv;
  }

  // ---- Phase 3: quantized probabilities -> PV WMMA (chunks ck == w mod 4) ----
  v8f o0 = {}, o1 = {}, o2 = {}, o3 = {};
  const int nchunk = (qt + 2) >> 1;                          // ceil((qt+1)/2)
  for (int ck = w; ck < nchunk; ck += 4) {
#pragma unroll
    for (int sub = 0; sub < 2; ++sub) {
      const int kt = ck * 2 + sub;
#pragma unroll
      for (int r = 0; r < 8; ++r) {
        float p = 0.0f;
        if (kt <= qt) {
          float sv = S[(mb + r) * TSEQ + kt * 16 + nn];
          // fake_quant(att): causal softmax => global min 0, max 1 exactly
          p = rintf(255.0f * __expf(sv - m[r]) * linv[r]) * (1.0f / 255.0f);
        }
        Pst[(mb + r) * 32 + sub * 16 + nn] = f2bf(p);
      }
    }
    asm volatile("s_wait_dscnt 0" ::: "memory");
    v16bf ap = load_a_frag(Pst, 32);                         // P tile 16x32
    const unsigned short* vb = vtb + (size_t)g * HS * TSEQ + ck * 32;
    o0 = wmma_bf16(ap, load_b_frag(vb + (size_t) 0 * TSEQ, TSEQ), o0);
    o1 = wmma_bf16(ap, load_b_frag(vb + (size_t)16 * TSEQ, TSEQ), o1);
    o2 = wmma_bf16(ap, load_b_frag(vb + (size_t)32 * TSEQ, TSEQ), o2);
    o3 = wmma_bf16(ap, load_b_frag(vb + (size_t)48 * TSEQ, TSEQ), o3);
  }

  // merge per-wave partial O tiles via LDS float atomics (ds_add_f32)
#pragma unroll
  for (int r = 0; r < 8; ++r) {
    atomicAdd(&obuf[(mb + r) * HS +  0 + nn], o0[r]);
    atomicAdd(&obuf[(mb + r) * HS + 16 + nn], o1[r]);
    atomicAdd(&obuf[(mb + r) * HS + 32 + nn], o2[r]);
    atomicAdd(&obuf[(mb + r) * HS + 48 + nn], o3[r]);
  }
  __syncthreads();

  // cooperative contiguous store of the 16x64 tile (global_store_b128)
  float4* dst = (float4*)(yf + ((size_t)h * TSEQ + qt * 16) * HS);
  const float4* srcb = (const float4*)obuf;
  dst[tid]       = srcb[tid];
  dst[tid + 128] = srcb[tid + 128];
}

// ---------------------------------------------------------------------------
extern "C" void kernel_launch(void* const* d_in, const int* in_sizes, int n_in,
                              void* d_out, int out_size, void* d_ws, size_t ws_size,
                              hipStream_t stream) {
  const float* x     = (const float*)d_in[0];
  const float* Wq    = (const float*)d_in[1];
  const float* Wk    = (const float*)d_in[2];
  const float* Wv    = (const float*)d_in[3];
  const float* Wproj = (const float*)d_in[4];
  const float* fw1   = (const float*)d_in[5];
  const float* fb1   = (const float*)d_in[6];
  const float* fw2   = (const float*)d_in[7];
  const float* fb2   = (const float*)d_in[8];
  const float* fc    = (const float*)d_in[9];
  const float* fL    = (const float*)d_in[10];
  float* out = (float*)d_out;

  char* ws = (char*)d_ws;
  size_t off = 0;
  auto alloc = [&](size_t bytes) -> void* {
    void* p = ws + off;
    off += (bytes + 255) & ~(size_t)255;
    return p;
  };
  unsigned short* xb  = (unsigned short*)alloc((size_t)TSEQ * CDIM * 2);
  unsigned short* Wqt = (unsigned short*)alloc((size_t)CDIM * CDIM * 2);
  unsigned short* Wkt = (unsigned short*)alloc((size_t)CDIM * KVDIM * 2);
  unsigned short* Wvt = (unsigned short*)alloc((size_t)CDIM * KVDIM * 2);
  unsigned short* Wpt = (unsigned short*)alloc((size_t)CDIM * CDIM * 2);
  float* qf = (float*)alloc((size_t)TSEQ * CDIM * 4);
  float* kf = (float*)alloc((size_t)TSEQ * KVDIM * 4);
  float* vf = (float*)alloc((size_t)TSEQ * KVDIM * 4);
  unsigned short* qbq = (unsigned short*)alloc((size_t)NHEAD * TSEQ * HS * 2);
  unsigned short* kbq = (unsigned short*)alloc((size_t)NKV * TSEQ * HS * 2);
  unsigned short* vtb = (unsigned short*)alloc((size_t)NKV * HS * TSEQ * 2);
  float* yf = (float*)alloc((size_t)NHEAD * TSEQ * HS * 4);
  unsigned short* yq = (unsigned short*)alloc((size_t)TSEQ * CDIM * 2);
  unsigned int* mm = (unsigned int*)alloc(8 * sizeof(unsigned int));

  const int thr = 256;
  f32_to_bf16<<<(TSEQ * CDIM + thr - 1) / thr, thr, 0, stream>>>(x, xb, TSEQ * CDIM);
  transpose_f32_to_bf16<<<(CDIM * CDIM + thr - 1) / thr, thr, 0, stream>>>(Wq, Wqt, CDIM, CDIM);
  transpose_f32_to_bf16<<<(CDIM * KVDIM + thr - 1) / thr, thr, 0, stream>>>(Wk, Wkt, CDIM, KVDIM);
  transpose_f32_to_bf16<<<(CDIM * KVDIM + thr - 1) / thr, thr, 0, stream>>>(Wv, Wvt, CDIM, KVDIM);
  transpose_f32_to_bf16<<<(CDIM * CDIM + thr - 1) / thr, thr, 0, stream>>>(Wproj, Wpt, CDIM, CDIM);

  auto gemm = [&](const unsigned short* A, const unsigned short* Bt, float* C,
                  int M, int N, int K) {
    int waves = (M / 16) * (N / 64);
    int blocks = (waves * 32 + thr - 1) / thr;
    gemm_bf16_nt<<<blocks, thr, 0, stream>>>(A, Bt, C, M, N, K);
  };
  gemm(xb, Wqt, qf, TSEQ, CDIM, CDIM);
  gemm(xb, Wkt, kf, TSEQ, KVDIM, CDIM);
  gemm(xb, Wvt, vf, TSEQ, KVDIM, CDIM);

  rope_kernel<<<(TSEQ * NHEAD * 32 + thr - 1) / thr, thr, 0, stream>>>(qf, NHEAD);
  rope_kernel<<<(TSEQ * NKV * 32 + thr - 1) / thr, thr, 0, stream>>>(kf, NKV);

  init_mm<<<1, 32, 0, stream>>>(mm);
  minmax_kernel<<<512, thr, 0, stream>>>(qf, TSEQ * CDIM, mm + 0);
  minmax_kernel<<<256, thr, 0, stream>>>(kf, TSEQ * KVDIM, mm + 2);
  minmax_kernel<<<256, thr, 0, stream>>>(vf, TSEQ * KVDIM, mm + 4);

  quant_qk<<<(TSEQ * CDIM + thr - 1) / thr, thr, 0, stream>>>(qf, qbq, mm, 0, NHEAD);
  quant_qk<<<(TSEQ * KVDIM + thr - 1) / thr, thr, 0, stream>>>(kf, kbq, mm, 2, NKV);
  quant_v<<<(TSEQ * KVDIM + thr - 1) / thr, thr, 0, stream>>>(vf, vtb, mm);

  attn_kernel<<<dim3(TSEQ / 16, NHEAD), 128, SMEM_TOTAL, stream>>>(
      qbq, kbq, vtb, yf, fw1, fb1, fw2, fb2, fc, fL);

  minmax_kernel<<<512, thr, 0, stream>>>(yf, NHEAD * TSEQ * HS, mm + 6);
  quant_y<<<(NHEAD * TSEQ * HS + thr - 1) / thr, thr, 0, stream>>>(yf, yq, mm);

  gemm(yq, Wpt, out, TSEQ, CDIM, CDIM);
}
